// SwinBlock_50182397887208
// MI455X (gfx1250) — compile-verified, matrix operand order checked by reading, not run on hardware
//
#include <hip/hip_runtime.h>
#include <cstdint>
#include <cstddef>

// ---------------------------------------------------------------------------
// CDNA5 (gfx1250) Swin block. bf16 WMMA (v_wmma_f32_16x16x32_bf16) for all
// GEMMs + window attention; async global->LDS staging of weight strips
// (global_load_async_to_lds_b128 + s_wait_asynccnt); fp32 residual stream.
// ---------------------------------------------------------------------------

typedef __bf16 bf16;
typedef __attribute__((ext_vector_type(8)))  bf16  v8bf;
typedef __attribute__((ext_vector_type(16))) bf16  v16bf;
typedef __attribute__((ext_vector_type(8)))  float v8f;

#define WSZ   7
#define NHD   6
#define HH    112
#define WW    112
#define CC    180
#define LL    (HH*WW)      // 12544 tokens per image
#define NB    2            // batch
#define MTOK  (NB*LL)      // 25088 rows in all token GEMMs
#define DH    30           // head dim
#define KP1   192          // C padded to K-multiple of 32
#define NQKV  576          // 3C padded to multiple of 64 (branch-free tiles)
#define NFC1  768          // 4C padded to multiple of 64 (also next GEMM's K)
#define ATT_SCALE 0.18257418583505536f   // 30^-0.5

__device__ __forceinline__ bf16 f2bf(float f) {
  union { float f; uint32_t u; } cf; cf.f = f;
  uint32_t u = cf.u + 0x7fffu + ((cf.u >> 16) & 1u);   // RNE
  union { uint16_t s; bf16 b; } cb; cb.s = (uint16_t)(u >> 16);
  return cb.b;
}

__device__ __forceinline__ float wred32(float v) {
#pragma unroll
  for (int m = 16; m > 0; m >>= 1) v += __shfl_xor(v, m, 32);
  return v;
}

// 16x32 bf16 fragment (A row-per-lane / B col-per-lane, K-major), ISA 7.12.2:
// lane<16: row=lane,    K = {k0..k0+7, k0+16..k0+23}
// lane>=16: row=lane-16, K = {k0+8..k0+15, k0+24..k0+31}
__device__ __forceinline__ v16bf load_frag(const bf16* base, int stride, int lane) {
  int r  = lane & 15;
  int kh = (lane >> 4) << 3;
  const bf16* p = base + (size_t)r * stride + kh;
  v8bf lo = *(const v8bf*)(p);
  v8bf hi = *(const v8bf*)(p + 16);
  v16bf a;
#pragma unroll
  for (int e = 0; e < 8; ++e) { a[e] = lo[e]; a[e + 8] = hi[e]; }
  return a;
}

__device__ __forceinline__ int regionid(int yp, int xp) {  // shift==3 regions
  int ry = (yp < 105) ? 0 : ((yp < 109) ? 1 : 2);
  int rx = (xp < 105) ? 0 : ((xp < 109) ? 1 : 2);
  return ry * 3 + rx;
}

// ------------------------- weight prep: (K,N) fp32 -> (Np,Kp) bf16 ----------
__global__ __launch_bounds__(256) void prep_w(
    const float* __restrict__ src, bf16* __restrict__ dst,
    int K, int N, int Kp, int Np) {
  int idx = blockIdx.x * 256 + threadIdx.x;
  if (idx >= Np * Kp) return;
  int n = idx / Kp, k = idx - n * Kp;
  float v = (n < N && k < K) ? src[(size_t)k * N + n] : 0.f;
  dst[idx] = f2bf(v);
}

// ------------------------- LayerNorms (one wave per token) ------------------
__global__ __launch_bounds__(256) void patch_ln(
    const float* __restrict__ x, const float* __restrict__ g,
    const float* __restrict__ bta, float* __restrict__ t) {
  int lane = threadIdx.x & 31;
  int tok = blockIdx.x * 8 + (threadIdx.x >> 5);
  int b = tok / LL, l = tok - b * LL;
  float v[6]; float s = 0.f, s2 = 0.f;
#pragma unroll
  for (int k = 0; k < 6; ++k) {
    int c = k * 32 + lane;
    float xv = (c < CC) ? x[((size_t)b * CC + c) * LL + l] : 0.f;
    v[k] = xv; s += xv; s2 += xv * xv;
  }
  s = wred32(s); s2 = wred32(s2);
  float mean = s * (1.f / CC);
  float rs = rsqrtf(s2 * (1.f / CC) - mean * mean + 1e-5f);
#pragma unroll
  for (int k = 0; k < 6; ++k) {
    int c = k * 32 + lane;
    if (c < CC) t[(size_t)tok * CC + c] = (v[k] - mean) * rs * g[c] + bta[c];
  }
}

__global__ __launch_bounds__(256) void ln_bf16(
    const float* __restrict__ t, const float* __restrict__ g,
    const float* __restrict__ bta, bf16* __restrict__ out, int Kp) {
  int lane = threadIdx.x & 31;
  int tok = blockIdx.x * 8 + (threadIdx.x >> 5);
  float v[6]; float s = 0.f, s2 = 0.f;
#pragma unroll
  for (int k = 0; k < 6; ++k) {
    int c = k * 32 + lane;
    float xv = (c < CC) ? t[(size_t)tok * CC + c] : 0.f;
    v[k] = xv; s += xv; s2 += xv * xv;
  }
  s = wred32(s); s2 = wred32(s2);
  float mean = s * (1.f / CC);
  float rs = rsqrtf(s2 * (1.f / CC) - mean * mean + 1e-5f);
#pragma unroll
  for (int k = 0; k < 6; ++k) {
    int c = k * 32 + lane;
    if (c < CC)      out[(size_t)tok * Kp + c] = f2bf((v[k] - mean) * rs * g[c] + bta[c]);
    else if (c < Kp) out[(size_t)tok * Kp + c] = f2bf(0.f);  // zero K-pad
  }
}

__global__ __launch_bounds__(256) void final_ln(
    const float* __restrict__ t, const float* __restrict__ g,
    const float* __restrict__ bta, float* __restrict__ out) {
  int lane = threadIdx.x & 31;
  int tok = blockIdx.x * 8 + (threadIdx.x >> 5);
  int b = tok / LL, l = tok - b * LL;
  float v[6]; float s = 0.f, s2 = 0.f;
#pragma unroll
  for (int k = 0; k < 6; ++k) {
    int c = k * 32 + lane;
    float xv = (c < CC) ? t[(size_t)tok * CC + c] : 0.f;
    v[k] = xv; s += xv; s2 += xv * xv;
  }
  s = wred32(s); s2 = wred32(s2);
  float mean = s * (1.f / CC);
  float rs = rsqrtf(s2 * (1.f / CC) - mean * mean + 1e-5f);
#pragma unroll
  for (int k = 0; k < 6; ++k) {
    int c = k * 32 + lane;
    if (c < CC) out[((size_t)b * CC + c) * LL + l] = (v[k] - mean) * rs * g[c] + bta[c];
  }
}

// ------------------------- bf16 WMMA GEMM with async-LDS B staging ----------
// Block = 4 waves; block tile 64(M) x 64(N). Each wave owns a 16x64 strip.
// Per K-step, the 64x32 bf16 weight strip (shared by all 4 waves) is staged
// through LDS with global_load_async_to_lds_b128, double-buffered on ASYNCcnt.
// A: (rows, KP) bf16 row-major.  W: (Np, KP) bf16 (pre-transposed, K-major),
// Np padded to a multiple of 64 so every tile is unconditionally valid.
// MODE 0: out bf16 (stride outStride), +bias, pad cols -> 0
// MODE 1: same + exact GELU
// MODE 2: out fp32 (stride outStride), out += acc + bias, cols < Nvalid only
template <int KP, int MODE>
__global__ __launch_bounds__(128) void gemm_wmma(
    const bf16* __restrict__ A, const bf16* __restrict__ W,
    const float* __restrict__ bias, void* __restrict__ outp,
    int Nvalid, int outStride) {
  constexpr int NSTEP = KP >> 5;
  __shared__ __align__(16) bf16 Wlds[2][64 * 32];

  int tid  = threadIdx.x;
  int lane = tid & 31;
  int wave = tid >> 5;
  int row0 = (blockIdx.x * 4 + wave) * 16;
  int col0 = blockIdx.y * 64;
  const bf16* Ab = A + (size_t)row0 * KP;
  const bf16* Wb = W + (size_t)col0 * KP;

  // Each thread async-copies two 16B chunks of the 64x32 strip per K-step.
  int c0 = tid * 2;
  int r0c = c0 >> 2, p0 = c0 & 3;
  int c1 = c0 + 1;
  int r1c = c1 >> 2, p1 = c1 & 3;

  auto issue = [&](int ks, int buf) {
    uint64_t g0 = (uint64_t)(uintptr_t)(Wb + (size_t)r0c * KP + ks * 32 + p0 * 8);
    uint64_t g1 = (uint64_t)(uintptr_t)(Wb + (size_t)r1c * KP + ks * 32 + p1 * 8);
    unsigned l0 = (unsigned)(uintptr_t)(&Wlds[buf][r0c * 32 + p0 * 8]);
    unsigned l1 = (unsigned)(uintptr_t)(&Wlds[buf][r1c * 32 + p1 * 8]);
    asm volatile(
        "global_load_async_to_lds_b128 %0, %1, off\n\t"
        "global_load_async_to_lds_b128 %2, %3, off"
        :: "v"(l0), "v"(g0), "v"(l1), "v"(g1) : "memory");
  };

  issue(0, 0);

  v8f acc0 = {0.f,0.f,0.f,0.f,0.f,0.f,0.f,0.f};
  v8f acc1 = acc0, acc2 = acc0, acc3 = acc0;

#pragma unroll
  for (int ks = 0; ks < NSTEP; ++ks) {
    int buf = ks & 1;
    if (ks + 1 < NSTEP) {
      issue(ks + 1, buf ^ 1);
      asm volatile("s_wait_asynccnt 0x2" ::: "memory");  // current buffer done
    } else {
      asm volatile("s_wait_asynccnt 0x0" ::: "memory");
    }
    __syncthreads();                                     // strip visible to WG
    v16bf a  = load_frag(Ab + ks * 32, KP, lane);
    v16bf b0 = load_frag(&Wlds[buf][ 0 * 32], 32, lane);
    v16bf b1 = load_frag(&Wlds[buf][16 * 32], 32, lane);
    v16bf b2 = load_frag(&Wlds[buf][32 * 32], 32, lane);
    v16bf b3 = load_frag(&Wlds[buf][48 * 32], 32, lane);
    acc0 = __builtin_amdgcn_wmma_f32_16x16x32_bf16(false, a, false, b0, (short)0, acc0, false, false);
    acc1 = __builtin_amdgcn_wmma_f32_16x16x32_bf16(false, a, false, b1, (short)0, acc1, false, false);
    acc2 = __builtin_amdgcn_wmma_f32_16x16x32_bf16(false, a, false, b2, (short)0, acc2, false, false);
    acc3 = __builtin_amdgcn_wmma_f32_16x16x32_bf16(false, a, false, b3, (short)0, acc3, false, false);
    __syncthreads();                                     // done reading buf
  }

  int n = lane & 15, mb = (lane >> 4) << 3;   // D layout, ISA 7.12.2
  v8f accs[4] = {acc0, acc1, acc2, acc3};
#pragma unroll
  for (int nt = 0; nt < 4; ++nt) {
    int col = col0 + nt * 16 + n;
#pragma unroll
    for (int j = 0; j < 8; ++j) {
      int r = row0 + mb + j;
      float v = accs[nt][j];
      if constexpr (MODE == 2) {
        if (col < Nvalid) {
          float* o = (float*)outp;
          o[(size_t)r * outStride + col] += v + bias[col];
        }
      } else {
        float u = 0.f;
        if (col < Nvalid) {
          u = v + bias[col];
          if constexpr (MODE == 1) u = 0.5f * u * (1.f + erff(u * 0.70710678118654752f));
        }
        ((bf16*)outp)[(size_t)r * outStride + col] = f2bf(u);
      }
    }
  }
}

// ------------------------- window attention ---------------------------------
// One block (4 waves) per window; loops 6 heads. LDS staging, WMMA for
// QK^T (16 tiles) and P@V (8 tiles x 2 K-steps). Shift/mask/bias analytic.
__global__ __launch_bounds__(128) void attn_kernel(
    const bf16* __restrict__ qkv, bf16* __restrict__ outb,
    const float* __restrict__ rpbL, int shift) {
  __shared__ __align__(16) bf16 qs[64 * 32];
  __shared__ __align__(16) bf16 ksm[64 * 32];
  __shared__ __align__(16) bf16 vT[32 * 64];
  __shared__ __align__(16) float sc[64 * 64];
  __shared__ __align__(16) bf16 pbm[64 * 64];
  __shared__ int tokL[49];

  int tid = threadIdx.x;
  int lane = tid & 31;
  int wave = tid >> 5;
  int w = blockIdx.x;
  int b = w >> 8;
  int wy = (w >> 4) & 15;
  int wx = w & 15;

  if (tid < 49) {  // shifted roll folded into the token gather index
    int iy = tid / 7, ix = tid % 7;
    int y = wy * 7 + iy + shift; if (y >= HH) y -= HH;
    int xg = wx * 7 + ix + shift; if (xg >= WW) xg -= WW;
    tokL[tid] = b * LL + y * WW + xg;
  }
  __syncthreads();

  const bf16 z = f2bf(0.f);
  v8f z8 = {0.f,0.f,0.f,0.f,0.f,0.f,0.f,0.f};

  for (int h = 0; h < NHD; ++h) {
    for (int idx = tid; idx < 64 * 32; idx += 128) {       // q, k (row, d)
      int i = idx >> 5, d = idx & 31;
      bf16 qv = z, kv = z;
      if (i < 49 && d < DH) {
        size_t base = (size_t)tokL[i] * NQKV + h * DH + d;
        qv = qkv[base];
        kv = qkv[base + 180];
      }
      qs[idx] = qv; ksm[idx] = kv;
    }
    for (int idx = tid; idx < 32 * 64; idx += 128) {       // v transposed (d, j)
      int d = idx >> 6, j = idx & 63;
      bf16 vv = z;
      if (j < 49 && d < DH) vv = qkv[(size_t)tokL[j] * NQKV + 360 + h * DH + d];
      vT[idx] = vv;
    }
    __syncthreads();

    for (int tix = wave; tix < 16; tix += 4) {             // QK^T
      int mt = tix >> 2, nt = tix & 3;
      v16bf a = load_frag(qs + mt * 16 * 32, 32, lane);
      v16bf bb = load_frag(ksm + nt * 16 * 32, 32, lane);
      v8f acc = __builtin_amdgcn_wmma_f32_16x16x32_bf16(
          false, a, false, bb, (short)0, z8, false, false);
      int n = lane & 15, mb = (lane >> 4) << 3;
#pragma unroll
      for (int j = 0; j < 8; ++j)
        sc[(mt * 16 + mb + j) * 64 + nt * 16 + n] = acc[j];
    }
    __syncthreads();

    if (tid < 64) {                                        // softmax per row
      int r = tid;
      if (r < 49) {
        int iy = r / 7, ix = r % 7;
        int cnti = shift ? regionid(wy * 7 + iy, wx * 7 + ix) : 0;
        float mx = -1e30f;
        for (int j = 0; j < 49; ++j) {
          int jy = j / 7, jx = j % 7;
          float bias = rpbL[((iy - jy + 6) * 13 + (ix - jx + 6)) * NHD + h];
          float msk = 0.f;
          if (shift) {
            int cntj = regionid(wy * 7 + jy, wx * 7 + jx);
            if (cntj != cnti) msk = -100.f;
          }
          float v = sc[r * 64 + j] * ATT_SCALE + bias + msk;
          sc[r * 64 + j] = v;
          mx = fmaxf(mx, v);
        }
        float s = 0.f;
        for (int j = 0; j < 49; ++j) {
          float e = __expf(sc[r * 64 + j] - mx);
          sc[r * 64 + j] = e; s += e;
        }
        float inv = 1.f / s;
        for (int j = 0; j < 64; ++j)
          pbm[r * 64 + j] = (j < 49) ? f2bf(sc[r * 64 + j] * inv) : z;
      } else {
        for (int j = 0; j < 64; ++j) pbm[r * 64 + j] = z;
      }
    }
    __syncthreads();

    for (int tix = wave * 2; tix < wave * 2 + 2; ++tix) {  // P @ V
      int mt = tix >> 1, nt = tix & 1;
      v8f acc = z8;
#pragma unroll
      for (int ks2 = 0; ks2 < 2; ++ks2) {
        v16bf a = load_frag(pbm + mt * 16 * 64 + ks2 * 32, 64, lane);
        v16bf bb = load_frag(vT + nt * 16 * 64 + ks2 * 32, 64, lane);
        acc = __builtin_amdgcn_wmma_f32_16x16x32_bf16(
            false, a, false, bb, (short)0, acc, false, false);
      }
      int n = lane & 15, mb = (lane >> 4) << 3;
      int d = nt * 16 + n;
#pragma unroll
      for (int j = 0; j < 8; ++j) {
        int i = mt * 16 + mb + j;
        if (i < 49 && d < DH)
          outb[(size_t)tokL[i] * KP1 + h * DH + d] = f2bf(acc[j]);
      }
    }
    __syncthreads();
  }
}

// ------------------------- orchestration ------------------------------------
extern "C" void kernel_launch(void* const* d_in, const int* in_sizes, int n_in,
                              void* d_out, int out_size, void* d_ws, size_t ws_size,
                              hipStream_t stream) {
  const float* x      = (const float*)d_in[0];
  const float* pe_g   = (const float*)d_in[1];
  const float* pe_b   = (const float*)d_in[2];
  const float* ln1_g  = (const float*)d_in[3];
  const float* ln1_b  = (const float*)d_in[4];
  const float* qkv_w  = (const float*)d_in[5];
  const float* qkv_b  = (const float*)d_in[6];
  const float* proj_w = (const float*)d_in[7];
  const float* proj_b = (const float*)d_in[8];
  const float* ln2_g  = (const float*)d_in[9];
  const float* ln2_b  = (const float*)d_in[10];
  const float* fc1_w  = (const float*)d_in[11];
  const float* fc1_b  = (const float*)d_in[12];
  const float* fc2_w  = (const float*)d_in[13];
  const float* fc2_b  = (const float*)d_in[14];
  const float* rpb    = (const float*)d_in[15];
  const float* fn_g   = (const float*)d_in[16];
  const float* fn_b   = (const float*)d_in[17];

  char* p = (char*)d_ws;
  auto alloc = [&](size_t n) { char* r = p; p += (n + 255) & ~(size_t)255; return r; };
  float* t    = (float*)alloc((size_t)MTOK * CC * 4);        // fp32 residual stream
  bf16* big   = (bf16*)alloc((size_t)MTOK * NFC1 * 2);       // qkv-out / fc1-out (aliased)
  bf16* abuf  = (bf16*)alloc((size_t)MTOK * KP1 * 2);        // LN-out / attn-out (aliased)
  bf16* wqkv  = (bf16*)alloc((size_t)4 * NQKV * KP1 * 2);
  bf16* wproj = (bf16*)alloc((size_t)4 * 192 * KP1 * 2);
  bf16* wfc1  = (bf16*)alloc((size_t)4 * NFC1 * KP1 * 2);
  bf16* wfc2  = (bf16*)alloc((size_t)4 * 192 * NFC1 * 2);

  for (int i = 0; i < 4; ++i) {  // bf16 transposed/padded weights
    prep_w<<<(NQKV * KP1 + 255) / 256, 256, 0, stream>>>(
        qkv_w + (size_t)i * CC * 540, wqkv + (size_t)i * NQKV * KP1, CC, 540, KP1, NQKV);
    prep_w<<<(192 * KP1 + 255) / 256, 256, 0, stream>>>(
        proj_w + (size_t)i * CC * CC, wproj + (size_t)i * 192 * KP1, CC, CC, KP1, 192);
    prep_w<<<(NFC1 * KP1 + 255) / 256, 256, 0, stream>>>(
        fc1_w + (size_t)i * CC * 720, wfc1 + (size_t)i * NFC1 * KP1, CC, 720, KP1, NFC1);
    prep_w<<<(192 * NFC1 + 255) / 256, 256, 0, stream>>>(
        fc2_w + (size_t)i * 720 * CC, wfc2 + (size_t)i * 192 * NFC1, 720, CC, NFC1, 192);
  }

  patch_ln<<<MTOK / 8, 256, 0, stream>>>(x, pe_g, pe_b, t);

  for (int i = 0; i < 4; ++i) {
    int shift = (i & 1) ? 3 : 0;
    ln_bf16<<<MTOK / 8, 256, 0, stream>>>(t, ln1_g + i * CC, ln1_b + i * CC, abuf, KP1);
    gemm_wmma<KP1, 0><<<dim3(MTOK / 64, NQKV / 64), 128, 0, stream>>>(
        abuf, wqkv + (size_t)i * NQKV * KP1, qkv_b + (size_t)i * 540,
        big, 540, NQKV);
    attn_kernel<<<512, 128, 0, stream>>>(big, abuf, rpb + (size_t)i * 169 * NHD, shift);
    gemm_wmma<KP1, 2><<<dim3(MTOK / 64, 192 / 64), 128, 0, stream>>>(
        abuf, wproj + (size_t)i * 192 * KP1, proj_b + (size_t)i * CC,
        t, CC, CC);
    ln_bf16<<<MTOK / 8, 256, 0, stream>>>(t, ln2_g + i * CC, ln2_b + i * CC, abuf, KP1);
    gemm_wmma<KP1, 1><<<dim3(MTOK / 64, NFC1 / 64), 128, 0, stream>>>(
        abuf, wfc1 + (size_t)i * NFC1 * KP1, fc1_b + (size_t)i * 720,
        big, 720, NFC1);
    gemm_wmma<NFC1, 2><<<dim3(MTOK / 64, 192 / 64), 128, 0, stream>>>(
        big, wfc2 + (size_t)i * 192 * NFC1, fc2_b + (size_t)i * CC,
        t, CC, CC);
  }

  final_ln<<<MTOK / 8, 256, 0, stream>>>(t, fn_g, fn_b, (float*)d_out);
}
